// CurricularFace_39505109189162
// MI455X (gfx1250) — matrix operand order repaired and checked by reading, not compiled, not run on hardware
//
#include <hip/hip_runtime.h>

#define DD 512
#define BB 512
#define CC 100000

#define COS_M_C  0.8775825618903728f
#define SIN_M_C  0.479425538604203f
#define THRESH_C (-0.8775825618903728f)
#define MM_C     0.2397127693021015f
#define S_C      64.0f

typedef __attribute__((ext_vector_type(16))) __bf16        v16bf;
typedef __attribute__((ext_vector_type(8)))  float         v8f;
typedef __attribute__((ext_vector_type(4)))  unsigned int  u32x4;
typedef __attribute__((ext_vector_type(4)))  float         f32x4;

__device__ __forceinline__ unsigned short f2bf(float x) {
    unsigned u = __float_as_uint(x);
    u += 0x7FFFu + ((u >> 16) & 1u);   // round-to-nearest-even
    return (unsigned short)(u >> 16);
}

// ---------------------------------------------------------------------------
// Kernel 1: L2-normalize embedding rows, emit bf16 A matrix [512][512]
// ---------------------------------------------------------------------------
__global__ __launch_bounds__(256)
void cf_norm_emb(const float* __restrict__ emb, unsigned short* __restrict__ Abf) {
    __shared__ float red[256];
    const int r = blockIdx.x, t = threadIdx.x;
    float e0 = emb[r * DD + t];
    float e1 = emb[r * DD + t + 256];
    red[t] = e0 * e0 + e1 * e1;
    __syncthreads();
    for (int s = 128; s > 0; s >>= 1) {
        if (t < s) red[t] += red[t + s];
        __syncthreads();
    }
    float inv = rsqrtf(red[0]);
    Abf[r * DD + t]       = f2bf(e0 * inv);
    Abf[r * DD + t + 256] = f2bf(e1 * inv);
}

// ---------------------------------------------------------------------------
// Kernel 2: per-row target logit, cos_theta_m, final target logit, EMA new_t
// ---------------------------------------------------------------------------
__global__ __launch_bounds__(512)
void cf_target(const float* __restrict__ emb, const int* __restrict__ labels,
               const float* __restrict__ Kmat, const float* __restrict__ tIn,
               float* __restrict__ ctm, float* __restrict__ ftl,
               float* __restrict__ newt) {
    __shared__ float red[512];
    const int b = threadIdx.x;
    const int lab = labels[b];
    float e2 = 0.f, k2 = 0.f, dt = 0.f;
    for (int d = 0; d < DD; ++d) {
        float e  = emb[b * DD + d];
        float kv = Kmat[(size_t)d * CC + lab];
        e2 += e * e; k2 += kv * kv; dt += e * kv;
    }
    float tgt = dt * rsqrtf(e2) * rsqrtf(k2);
    tgt = fminf(fmaxf(tgt, -1.f), 1.f);
    float sn = sqrtf(fmaxf(1.f - tgt * tgt, 0.f));
    float cm = tgt * COS_M_C - sn * SIN_M_C;
    ctm[b] = cm;
    ftl[b] = (tgt > THRESH_C) ? cm : (tgt - MM_C);
    red[b] = tgt;
    __syncthreads();
    for (int s = 256; s > 0; s >>= 1) {
        if (b < s) red[b] += red[b + s];
        __syncthreads();
    }
    if (b == 0) newt[0] = 0.01f * (red[0] * (1.f / 512.f)) + 0.99f * tIn[0];
}

// ---------------------------------------------------------------------------
// Kernel 3: software-pipelined WMMA bf16 GEMM (128x128 tile, K-step 32),
// column norms fused into staging, CurricularFace epilogue, NT stores.
// ---------------------------------------------------------------------------
__global__ __launch_bounds__(256)
void cf_gemm(const unsigned short* __restrict__ Abf,
             const float* __restrict__ Kmat,
             const int* __restrict__ labels,
             const float* __restrict__ ctm,
             const float* __restrict__ ftl,
             const float* __restrict__ pnewt,
             float* __restrict__ out) {
    __shared__ __align__(16) unsigned short Atile[128 * 40];  // [M][K] padded
    __shared__ __align__(16) unsigned short Btile[128 * 40];  // [N][K] padded
    __shared__ float colSS[128];
    __shared__ float sCtm[128];
    __shared__ float sFtl[128];
    __shared__ int   sLab[128];

    const int tid  = threadIdx.x;
    const int lane = tid & 31;
    const int wave = tid >> 5;
    const int wm   = wave & 1;   // 2 waves along M (64 rows each)
    const int wn   = wave >> 1;  // 4 waves along N (32 cols each)
    const int mBase = blockIdx.x * 128;
    const int nBase = blockIdx.y * 128;

    if (tid < 128) {
        colSS[tid] = 0.f;
        int r = mBase + tid;
        sCtm[tid] = ctm[r];
        sFtl[tid] = ftl[r];
        sLab[tid] = labels[r];
    }

    // --- per-thread staging geometry (fixed across k-tiles) ---
    // A: two 8-half chunks, rows rA and rA+64, k-offset kAoff
    const int rA    = tid >> 2;
    const int kAoff = (tid & 3) * 8;
    const unsigned short* aSrc0 = Abf + (size_t)(mBase + rA) * DD + kAoff;
    const unsigned short* aSrc1 = aSrc0 + (size_t)64 * DD;
    unsigned short* aDst0 = &Atile[rA * 40 + kAoff];
    unsigned short* aDst1 = &Atile[(rA + 64) * 40 + kAoff];
    // B: 4 fixed columns (ngB..+3), k-rows kkB, kkB+8, kkB+16, kkB+24
    const int kkB = tid >> 5;
    const int ngB = (tid & 31) * 4;
    const int colB  = nBase + ngB;
    const bool colOK = colB < CC;
    const int colBc = colOK ? colB : (CC - 4);        // clamped, always valid
    const float* bSrcBase = Kmat + colBc;

    f32x4 ssq = {0.f, 0.f, 0.f, 0.f};

    v8f acc[4][2];
#pragma unroll
    for (int ms = 0; ms < 4; ++ms)
#pragma unroll
        for (int ns = 0; ns < 2; ++ns)
            acc[ms][ns] = (v8f){0.f, 0.f, 0.f, 0.f, 0.f, 0.f, 0.f, 0.f};

    const int rowL = lane & 15;
    const int hiA  = (lane >> 4) * 16;  // A k-base per half-wave
    const int hiB  = (lane >> 4) * 16;  // B k-base per half-wave

    // --- prime the pipeline: prefetch tile 0 into registers ---
    u32x4 pa0, pa1;
    f32x4 pb[4];
    {
        pa0 = *(const u32x4*)(aSrc0);
        pa1 = *(const u32x4*)(aSrc1);
#pragma unroll
        for (int i = 0; i < 4; ++i) {
            f32x4 v = *(const f32x4*)(bSrcBase + (size_t)(kkB + i * 8) * CC);
            pb[i].x = colOK ? v.x : 0.f;
            pb[i].y = colOK ? v.y : 0.f;
            pb[i].z = colOK ? v.z : 0.f;
            pb[i].w = colOK ? v.w : 0.f;
        }
    }

    for (int kt = 0; kt < 16; ++kt) {
        __syncthreads();   // previous tile's compute done; LDS free
        // ---- store prefetched tile to LDS (B: f32->bf16 + sumsq) ----
        *(u32x4*)aDst0 = pa0;
        *(u32x4*)aDst1 = pa1;
#pragma unroll
        for (int i = 0; i < 4; ++i) {
            int kk = kkB + i * 8;
            f32x4 v = pb[i];
            ssq += v * v;
            Btile[(ngB + 0) * 40 + kk] = f2bf(v.x);
            Btile[(ngB + 1) * 40 + kk] = f2bf(v.y);
            Btile[(ngB + 2) * 40 + kk] = f2bf(v.z);
            Btile[(ngB + 3) * 40 + kk] = f2bf(v.w);
        }
        __syncthreads();   // tile visible to all waves

        // ---- issue next tile's global loads (waits land after WMMAs) ----
        {
            int ktn = (kt < 15) ? (kt + 1) : 15;   // clamped: branch-free, in-bounds
            int k0n = ktn * 32;
            pa0 = *(const u32x4*)(aSrc0 + k0n);
            pa1 = *(const u32x4*)(aSrc1 + k0n);
#pragma unroll
            for (int i = 0; i < 4; ++i) {
                f32x4 v = *(const f32x4*)(bSrcBase + (size_t)(k0n + kkB + i * 8) * CC);
                pb[i].x = colOK ? v.x : 0.f;
                pb[i].y = colOK ? v.y : 0.f;
                pb[i].z = colOK ? v.z : 0.f;
                pb[i].w = colOK ? v.w : 0.f;
            }
        }

        // ---- WMMA: 4(M) x 2(N) 16x16x32 per wave ----
        union Frag { u32x4 q[2]; v16bf v; } af, bf[2];
#pragma unroll
        for (int ns = 0; ns < 2; ++ns) {
            const unsigned short* bp = &Btile[(wn * 32 + ns * 16 + rowL) * 40 + hiB];
            bf[ns].q[0] = *(const u32x4*)(bp);
            bf[ns].q[1] = *(const u32x4*)(bp + 8);
        }
#pragma unroll
        for (int ms = 0; ms < 4; ++ms) {
            const unsigned short* ap = &Atile[(wm * 64 + ms * 16 + rowL) * 40 + hiA];
            af.q[0] = *(const u32x4*)(ap);
            af.q[1] = *(const u32x4*)(ap + 8);
#pragma unroll
            for (int ns = 0; ns < 2; ++ns) {
                acc[ms][ns] = __builtin_amdgcn_wmma_f32_16x16x32_bf16(
                    false, af.v, false, bf[ns].v, (short)0, acc[ms][ns],
                    false, false);
            }
        }
    }

    // ---- reduce per-column sum of squares (8 partial threads per column) ----
    __syncthreads();
    atomicAdd(&colSS[ngB + 0], ssq.x);
    atomicAdd(&colSS[ngB + 1], ssq.y);
    atomicAdd(&colSS[ngB + 2], ssq.z);
    atomicAdd(&colSS[ngB + 3], ssq.w);
    __syncthreads();

    // ---- CurricularFace epilogue ----
    const float newt = pnewt[0];
#pragma unroll
    for (int ns = 0; ns < 2; ++ns) {
        int colLocal = wn * 32 + ns * 16 + rowL;
        int col = nBase + colLocal;
        if (col >= CC) continue;   // wave-uniform in the tail block
        float invn = rsqrtf(colSS[colLocal]);
#pragma unroll
        for (int ms = 0; ms < 4; ++ms) {
#pragma unroll
            for (int i = 0; i < 8; ++i) {
                int rloc = wm * 64 + ms * 16 + i + ((lane >> 4) * 8);
                int row  = mBase + rloc;
                float cosv = acc[ms][ns][i] * invn;
                cosv = fminf(fmaxf(cosv, -1.f), 1.f);
                float hard = cosv * (newt + cosv);
                float o = (cosv > sCtm[rloc]) ? hard : cosv;
                o = (col == sLab[rloc]) ? sFtl[rloc] : o;
                __builtin_nontemporal_store(o * S_C, out + (size_t)row * CC + col);
            }
        }
    }
}

// ---------------------------------------------------------------------------
extern "C" void kernel_launch(void* const* d_in, const int* in_sizes, int n_in,
                              void* d_out, int out_size, void* d_ws, size_t ws_size,
                              hipStream_t stream) {
    const float* emb    = (const float*)d_in[0];  // [512,512]
    const int*   labels = (const int*)d_in[1];    // [512]
    const float* Kmat   = (const float*)d_in[2];  // [512,100000]
    const float* tIn    = (const float*)d_in[3];  // scalar
    float* out = (float*)d_out;

    char* ws = (char*)d_ws;
    unsigned short* Abf = (unsigned short*)ws;              // 512*512*2 = 512 KB
    float* ctm  = (float*)(ws + (size_t)BB * DD * 2);       // 512 f32
    float* ftl  = ctm + BB;                                 // 512 f32
    float* newt = ftl + BB;                                 // 1 f32

    cf_norm_emb<<<BB, 256, 0, stream>>>(emb, Abf);
    cf_target<<<1, BB, 0, stream>>>(emb, labels, Kmat, tIn, ctm, ftl, newt);
    cf_gemm<<<dim3(4, (CC + 127) / 128), 256, 0, stream>>>(
        Abf, Kmat, labels, ctm, ftl, newt, out);
}
